// Attention_3813930959067
// MI455X (gfx1250) — matrix operand order repaired and checked by reading, not compiled, not run on hardware
//
#include <hip/hip_runtime.h>
#include <hip/hip_bf16.h>

typedef __attribute__((ext_vector_type(16))) __bf16 v16bf;
typedef __attribute__((ext_vector_type(8)))  __bf16 v8bf;
typedef __attribute__((ext_vector_type(4)))  __bf16 v4bf;
typedef __attribute__((ext_vector_type(8)))  float  v8f;

#define DIM      2560
#define NHEADS   40
#define HDIM     64
#define BATCH    2
#define SEQ      2048
#define ROWS     (BATCH*SEQ)          // 4096
#define C3       (3*DIM)              // 7680
#define QK_SCALE 0.125f               // 64^-0.5

// ---------------------------------------------------------------------------
// WMMA helper (CDNA5 16x16x32 bf16, fp32 accumulate)
// ---------------------------------------------------------------------------
__device__ __forceinline__ v8f wmma_bf16(v16bf a, v16bf b, v8f c) {
    return __builtin_amdgcn_wmma_f32_16x16x32_bf16(
        /*neg_a=*/false, a, /*neg_b=*/false, b,
        /*c_mod=*/(short)0, c, /*reuse_a=*/false, /*reuse_b=*/false);
}

// A-matrix fragment (16x32, row M = lane%16).
// half = lane>>4; elems 0..7 -> K = half*8 + e ; elems 8..15 -> K = 16 + half*8 + e
__device__ __forceinline__ v16bf load_a_frag(const __bf16* rowk, int half) {
    v8bf c1 = *(const v8bf*)(rowk + half * 8);
    v8bf c2 = *(const v8bf*)(rowk + 16 + half * 8);
    return __builtin_shufflevector(c1, c2, 0,1,2,3,4,5,6,7,8,9,10,11,12,13,14,15);
}
// B-matrix fragment (32x16, col N = lane%16), source stored as [N][K] row-major:
// elems 0..7 -> K = half*8 + e ; elems 8..15 -> K = 16 + half*8 + e
__device__ __forceinline__ v16bf load_b_frag(const __bf16* rowk, int half) {
    v8bf c1 = *(const v8bf*)(rowk + half * 8);
    v8bf c2 = *(const v8bf*)(rowk + 16 + half * 8);
    return __builtin_shufflevector(c1, c2, 0,1,2,3,4,5,6,7,8,9,10,11,12,13,14,15);
}

// ---------------------------------------------------------------------------
// Cross-lane butterfly reductions, single fused VOP2+DPP16 per step.
// Masks {1,2,7,15} are linearly independent over GF(2): span = all 16 lanes,
// never crossing the 16-lane halves (matches WMMA C-layout row striping).
// Exact for max (idempotent) and sum (disjoint cosets each step).
// Inputs are finite, so skipping fmaxf's sNaN canonicalize is safe.
// ---------------------------------------------------------------------------
#define MAKE_DPP_RED(name, mnem, ctrl)                                          \
__device__ __forceinline__ float name(float v) {                                \
    float d;                                                                    \
    asm(mnem " %0, %1, %1 " ctrl " row_mask:0xf bank_mask:0xf bound_ctrl:1"     \
        : "=v"(d) : "v"(v));                                                    \
    return d;                                                                   \
}
MAKE_DPP_RED(max_x1,  "v_max_num_f32_dpp", "quad_perm:[1,0,3,2]")
MAKE_DPP_RED(max_x2,  "v_max_num_f32_dpp", "quad_perm:[2,3,0,1]")
MAKE_DPP_RED(max_x7,  "v_max_num_f32_dpp", "row_half_mirror")
MAKE_DPP_RED(max_x15, "v_max_num_f32_dpp", "row_mirror")
MAKE_DPP_RED(add_x1,  "v_add_f32_dpp",     "quad_perm:[1,0,3,2]")
MAKE_DPP_RED(add_x2,  "v_add_f32_dpp",     "quad_perm:[2,3,0,1]")
MAKE_DPP_RED(add_x7,  "v_add_f32_dpp",     "row_half_mirror")
MAKE_DPP_RED(add_x15, "v_add_f32_dpp",     "row_mirror")

__device__ __forceinline__ float redmax16(float v) {
    v = max_x1(v); v = max_x2(v); v = max_x7(v); v = max_x15(v);
    return v;
}
__device__ __forceinline__ float redsum16(float v) {
    v = add_x1(v); v = add_x2(v); v = add_x7(v); v = add_x15(v);
    return v;
}

// ---------------------------------------------------------------------------
// fp32 -> bf16 convert (vectorized x4)
// ---------------------------------------------------------------------------
__global__ void cvt4_kernel(const float* __restrict__ in, __bf16* __restrict__ out, long n) {
    long i = ((long)blockIdx.x * blockDim.x + threadIdx.x) * 4;
    if (i < n) {
        float4 f = *(const float4*)(in + i);
        v4bf o;
        o[0] = (__bf16)f.x; o[1] = (__bf16)f.y; o[2] = (__bf16)f.z; o[3] = (__bf16)f.w;
        *(v4bf*)(out + i) = o;
    }
}

// transpose + convert: out[c*rows + k] = (bf16) in[k*cols + c]
__global__ void transpose_cvt_kernel(const float* __restrict__ in, __bf16* __restrict__ out,
                                     int rows /*K*/, int cols /*Cout*/) {
    __shared__ float tile[16][17];
    int tx = threadIdx.x & 15, ty = threadIdx.x >> 4;
    long c0 = (long)blockIdx.x * 16, k0 = (long)blockIdx.y * 16;
    tile[ty][tx] = in[(k0 + ty) * cols + c0 + tx];
    __syncthreads();
    out[(c0 + ty) * rows + k0 + tx] = (__bf16)tile[tx][ty];
}

// ---------------------------------------------------------------------------
// GEMM: Cacc[r,c] = sum_k A[r,k] * WT[c,k]   (A, WT bf16 row-major; fp32 acc)
// 256 threads = 8 waves (2 M x 4 N), wave tile 32x64 (2x4 WMMA accs),
// block tile 64x256.  8 WMMA : 12 b128-loads per K=32 step.
// EPI 0: QKV epilogue (bias, fold scale into Q, scatter q/k/vT)
// EPI 1: proj epilogue (bias, fp32 out)
// ---------------------------------------------------------------------------
template <int EPI, int COUT>
__global__ void gemm_kernel(const __bf16* __restrict__ A, const __bf16* __restrict__ WT,
                            const float* __restrict__ bias,
                            __bf16* __restrict__ qb, __bf16* __restrict__ kb,
                            __bf16* __restrict__ vT, float* __restrict__ outf) {
    constexpr int K = DIM;
    const int lane = threadIdx.x & 31;
    const int wave = threadIdx.x >> 5;
    const int ln16 = lane & 15, half = lane >> 4;
    const int wm = wave >> 2, wn = wave & 3;
    const long mbase = (long)blockIdx.y * 64 + wm * 32;
    const long nbase = (long)blockIdx.x * 256 + wn * 64;

    v8f acc[2][4] = {};

    const __bf16* arow0 = A + (mbase + ln16) * K;
    const __bf16* arow1 = arow0 + 16L * K;
    const __bf16* brow[4];
#pragma unroll
    for (int j = 0; j < 4; j++) brow[j] = WT + (nbase + j * 16 + ln16) * (long)K;

#pragma unroll 2
    for (int k = 0; k < K; k += 32) {
        __builtin_prefetch(arow0 + k + 256, 0, 1);   // global_prefetch_b8
        v16bf a0 = load_a_frag(arow0 + k, half);
        v16bf a1 = load_a_frag(arow1 + k, half);
#pragma unroll
        for (int j = 0; j < 4; j++) {
            v16bf bf = load_b_frag(brow[j] + k, half);
            acc[0][j] = wmma_bf16(a0, bf, acc[0][j]);
            acc[1][j] = wmma_bf16(a1, bf, acc[1][j]);
        }
    }

#pragma unroll
    for (int mt = 0; mt < 2; mt++) {
#pragma unroll
        for (int nt = 0; nt < 4; nt++) {
            const long cc = nbase + nt * 16 + ln16;
            const float bv = bias[cc];
            if (EPI == 0) {
                const int t = (int)(cc / DIM);
                const int rem = (int)(cc % DIM);
                const int h = rem >> 6, d = rem & 63;
#pragma unroll
                for (int r = 0; r < 8; r++) {
                    const long m = mbase + mt * 16 + half * 8 + r;
                    const long b = m >> 11, n = m & 2047;  // SEQ = 2048
                    float v = acc[mt][nt][r] + bv;
                    if (t == 0)      qb[(((b * NHEADS + h) * SEQ + n) << 6) + d] = (__bf16)(v * QK_SCALE);
                    else if (t == 1) kb[(((b * NHEADS + h) * SEQ + n) << 6) + d] = (__bf16)v;
                    else             vT[(((b * NHEADS + h) << 6) + d) * SEQ + n] = (__bf16)v;
                }
            } else {
#pragma unroll
                for (int r = 0; r < 8; r++) {
                    const long m = mbase + mt * 16 + half * 8 + r;
                    outf[m * COUT + cc] = acc[mt][nt][r] + bv;
                }
            }
        }
    }
}

// ---------------------------------------------------------------------------
// Flash attention: one wave = 16 query rows, 8 waves/block = 128 rows.
// grid = B*H*(SEQ/128) = 1280 blocks. Key blocks of 32 with online softmax.
// Per-wave-private LDS P tile: same-wave LDS ops are in-order (DScnt),
// so no workgroup barriers. All K/V fragment loads are issued at the top of
// the loop so the softmax VALU chain hides their latency.
// ---------------------------------------------------------------------------
__global__ void attn_kernel(const __bf16* __restrict__ qb, const __bf16* __restrict__ kb,
                            const __bf16* __restrict__ vT, __bf16* __restrict__ aout) {
    __shared__ __attribute__((aligned(32))) __bf16 Pst[8][16 * 32];  // per-wave P tile

    const int lane = threadIdx.x & 31;
    const int wave = threadIdx.x >> 5;
    const int ln16 = lane & 15, half = lane >> 4;

    const int qblk = blockIdx.x & 15;
    const int bh   = blockIdx.x >> 4;           // b*NHEADS + h
    const int b    = bh / NHEADS, h = bh % NHEADS;
    const long q0  = (long)qblk * 128 + wave * 16;

    const __bf16* qbase = qb + (long)bh * SEQ * HDIM;
    const __bf16* kbase = kb + (long)bh * SEQ * HDIM;
    const __bf16* vbase = vT + (long)bh * HDIM * SEQ;

    // Q fragments: A layout, K-windows d=[0,32) and d=[32,64)
    const __bf16* qrow = qbase + (q0 + ln16) * HDIM;
    const v16bf qa0 = load_a_frag(qrow + 0, half);
    const v16bf qa1 = load_a_frag(qrow + 32, half);

    v8f o[4] = {};
    float mrow[8], lrow[8];
#pragma unroll
    for (int r = 0; r < 8; r++) { mrow[r] = -1e30f; lrow[r] = 0.f; }

    __bf16* pw = Pst[wave];
    const __bf16* vrow0 = vbase + (long)ln16 * SEQ;

    for (int kb0 = 0; kb0 < SEQ; kb0 += 32) {
        // ---- issue ALL global fragment loads for this block up front:
        //      4 K-frags (S operands) + 4 V-frags (AV operands, needed only
        //      after the softmax -> fully latency-hidden)
        const __bf16* krow0 = kbase + (long)(kb0 + ln16) * HDIM;
        const __bf16* krow1 = krow0 + 16 * HDIM;
        __builtin_prefetch(krow0 + 32 * HDIM, 0, 1);     // next key block
        v16bf kf00 = load_b_frag(krow0 + 0, half);
        v16bf kf01 = load_b_frag(krow0 + 32, half);
        v16bf kf10 = load_b_frag(krow1 + 0, half);
        v16bf kf11 = load_b_frag(krow1 + 32, half);
        v16bf vf0  = load_b_frag(vrow0 + 0L * 16 * SEQ + kb0, half);
        v16bf vf1  = load_b_frag(vrow0 + 1L * 16 * SEQ + kb0, half);
        v16bf vf2  = load_b_frag(vrow0 + 2L * 16 * SEQ + kb0, half);
        v16bf vf3  = load_b_frag(vrow0 + 3L * 16 * SEQ + kb0, half);

        // ---- S = Q * K^T for 32 keys (two 16-key N-tiles, two K=32 d-windows)
        v8f s0 = {}, s1 = {};
        s0 = wmma_bf16(qa0, kf00, s0);
        s0 = wmma_bf16(qa1, kf01, s0);
        s1 = wmma_bf16(qa0, kf10, s1);
        s1 = wmma_bf16(qa1, kf11, s1);

        // ---- online softmax (rows m = half*8 + r, striped over 16-lane halves)
        float alpha[8];
#pragma unroll
        for (int r = 0; r < 8; r++) {
            float mx = redmax16(fmaxf(s0[r], s1[r]));
            float mnew = fmaxf(mrow[r], mx);
            alpha[r] = __expf(mrow[r] - mnew);
            mrow[r] = mnew;
            float p0 = __expf(s0[r] - mnew);
            float p1 = __expf(s1[r] - mnew);
            s0[r] = p0; s1[r] = p1;
            lrow[r] = lrow[r] * alpha[r] + redsum16(p0 + p1);
        }
#pragma unroll
        for (int j = 0; j < 4; j++)
#pragma unroll
            for (int r = 0; r < 8; r++) o[j][r] *= alpha[r];

        // ---- stage P (C layout -> LDS [row][key]) then reload as A fragment;
        //      same-wave LDS ordering makes this barrier-free
#pragma unroll
        for (int r = 0; r < 8; r++) {
            const int row = half * 8 + r;
            pw[row * 32 + ln16]      = (__bf16)s0[r];
            pw[row * 32 + 16 + ln16] = (__bf16)s1[r];
        }
        v16bf pa = load_a_frag(pw + ln16 * 32, half);

        // ---- O += P * V  (V stored transposed [d][n]: contiguous B frags)
        o[0] = wmma_bf16(pa, vf0, o[0]);
        o[1] = wmma_bf16(pa, vf1, o[1]);
        o[2] = wmma_bf16(pa, vf2, o[2]);
        o[3] = wmma_bf16(pa, vf3, o[3]);
    }

    // ---- epilogue: normalize, store to [B*N, C] bf16
#pragma unroll
    for (int r = 0; r < 8; r++) {
        const float inv = 1.0f / lrow[r];
        const long m = q0 + half * 8 + r;
        const long row = (long)b * SEQ + m;
#pragma unroll
        for (int j = 0; j < 4; j++) {
            const int col = h * HDIM + j * 16 + ln16;
            aout[row * DIM + col] = (__bf16)(o[j][r] * inv);
        }
    }
}

// ---------------------------------------------------------------------------
// Host-side launcher
// ---------------------------------------------------------------------------
extern "C" void kernel_launch(void* const* d_in, const int* in_sizes, int n_in,
                              void* d_out, int out_size, void* d_ws, size_t ws_size,
                              hipStream_t stream) {
    const float* x      = (const float*)d_in[0];
    const float* w_qkv  = (const float*)d_in[1];
    const float* b_qkv  = (const float*)d_in[2];
    const float* w_proj = (const float*)d_in[3];
    const float* b_proj = (const float*)d_in[4];
    float* out = (float*)d_out;

    char* ws = (char*)d_ws;
    const size_t SZ_X   = (size_t)ROWS * DIM * 2;      // 20.97 MB (also attn out)
    const size_t SZ_WQ  = (size_t)C3 * DIM * 2;        // 39.32 MB
    const size_t SZ_WP  = (size_t)DIM * DIM * 2;       // 13.11 MB
    const size_t SZ_QKV = (size_t)BATCH * NHEADS * SEQ * HDIM * 2;  // 20.97 MB each

    __bf16* xbf    = (__bf16*)(ws);
    __bf16* wqkvT  = (__bf16*)(ws + SZ_X);
    __bf16* wprojT = (__bf16*)(ws + SZ_X + SZ_WQ);
    __bf16* qbuf   = (__bf16*)(ws + SZ_X + SZ_WQ + SZ_WP);
    __bf16* kbuf   = (__bf16*)(ws + SZ_X + SZ_WQ + SZ_WP + SZ_QKV);
    __bf16* vTbuf  = (__bf16*)(ws + SZ_X + SZ_WQ + SZ_WP + 2 * SZ_QKV);
    __bf16* attnout = xbf;  // x no longer needed after QKV GEMM

    // 1) converts
    {
        long n = (long)ROWS * DIM;
        cvt4_kernel<<<(unsigned)(n / (256 * 4)), 256, 0, stream>>>(x, xbf, n);
        transpose_cvt_kernel<<<dim3(C3 / 16, DIM / 16), 256, 0, stream>>>(w_qkv, wqkvT, DIM, C3);
        transpose_cvt_kernel<<<dim3(DIM / 16, DIM / 16), 256, 0, stream>>>(w_proj, wprojT, DIM, DIM);
    }
    // 2) QKV GEMM + scatter (scale folded into q, v transposed)
    gemm_kernel<0, C3><<<dim3(C3 / 256, ROWS / 64), 256, 0, stream>>>(
        xbf, wqkvT, b_qkv, qbuf, kbuf, vTbuf, nullptr);
    // 3) flash attention
    attn_kernel<<<BATCH * NHEADS * (SEQ / 128), 256, 0, stream>>>(qbuf, kbuf, vTbuf, attnout);
    // 4) projection GEMM -> fp32 out
    gemm_kernel<1, DIM><<<dim3(DIM / 256, ROWS / 64), 256, 0, stream>>>(
        attnout, wprojT, b_proj, nullptr, nullptr, nullptr, out);
}
